// DeepDFText0_MessageAndUpdate_2619930050959
// MI455X (gfx1250) — matrix operand order repaired and checked by reading, not compiled, not run on hardware
//
#include <hip/hip_runtime.h>
#include <math.h>

// MI455X / gfx1250, wave32. D = A(16x32 f16) x B(32x16 f16) + C, f32 accum.
typedef __attribute__((ext_vector_type(16))) _Float16 v16h;
typedef __attribute__((ext_vector_type(8)))  float    v8f;

#define N_NODES 1024
#define F_DIM   32
#define E_DIM   20   // K padded to 32; K=20 carries the (mask-scaled) bias channel
#define IBLK    2    // i-rows per block: amortizes j-indexed loads across i

__device__ __forceinline__ float silu_f(float x) { return x / (1.f + expf(-x)); }

// ---------------------------------------------------------------------------
// Kernel 1: left[j,0:96] = silu(scalar[j] @ mL_w1 + b1) @ mL_w2 + b2
//           Q[j,d,f]     = left[j,f] * vector[j,d,f]   (i-independent prefold)
// ---------------------------------------------------------------------------
__global__ __launch_bounds__(128)
void k_msg_left(const float* __restrict__ scalar,
                const float* __restrict__ vector,
                const float* __restrict__ w1, const float* __restrict__ b1,
                const float* __restrict__ w2, const float* __restrict__ b2,
                float* __restrict__ leftbuf, float* __restrict__ Qbuf)
{
    const int j = blockIdx.x;
    const int tid = threadIdx.x;
    __shared__ float sin_[F_DIM];
    __shared__ float h[F_DIM];
    __shared__ float lout[3 * F_DIM];
    if (tid < F_DIM) sin_[tid] = scalar[j * F_DIM + tid];
    __syncthreads();
    if (tid < F_DIM) {
        float acc = b1[tid];
        #pragma unroll
        for (int k = 0; k < F_DIM; ++k) acc += sin_[k] * w1[k * F_DIM + tid];
        h[tid] = silu_f(acc);
    }
    __syncthreads();
    if (tid < 3 * F_DIM) {
        float acc = b2[tid];
        #pragma unroll
        for (int k = 0; k < F_DIM; ++k) acc += h[k] * w2[k * 3 * F_DIM + tid];
        lout[tid] = acc;
        leftbuf[j * 3 * F_DIM + tid] = acc;
    }
    __syncthreads();
    if (tid < 3 * F_DIM)   // tid = d*32 + f
        Qbuf[j * 3 * F_DIM + tid] = lout[tid & 31] * vector[j * 3 * F_DIM + tid];
}

// ---------------------------------------------------------------------------
// Kernel 2: message pass. One block per IBLK i-rows; 8 waves, each wave owns
// 8 j-tiles of 16. B = f16(mask[i,j] * expansion[i,j,0:20]) + mask bias
// channel at K=20, so D = mask-scaled right directly.
// D lane layout: col N = lane%16 -> fixed j per lane; row M = r + 8*(lane/16)
// -> 8 contiguous features per D register, so Q/left loads are float4.
// ---------------------------------------------------------------------------
__global__ __launch_bounds__(256)
void k_message(const float* __restrict__ scalar,
               const float* __restrict__ vector,
               const float* __restrict__ expansion,
               const float* __restrict__ direction,
               const float* __restrict__ mask,
               const float* __restrict__ mR_w,
               const float* __restrict__ mR_b,
               const float* __restrict__ leftbuf,
               const float* __restrict__ Qbuf,
               float* __restrict__ out_s,
               float* __restrict__ out_v)
{
    const int i0   = blockIdx.x * IBLK;
    const int tid  = threadIdx.x;
    const int lane = tid & 31;
    const int wv   = tid >> 5;        // wave 0..7
    const int n    = lane & 15;       // column (j within tile)
    const int hi   = lane >> 4;       // lane half

    __shared__ float sS[IBLK][F_DIM];
    __shared__ float sV[IBLK][3 * F_DIM];
    if (tid < IBLK * F_DIM)     ((float*)sS)[tid] = 0.f;
    if (tid < IBLK * 3 * F_DIM) ((float*)sV)[tid] = 0.f;
    __syncthreads();

    // --- A tiles: A[M=f_in_tile, K=e] = mR_w[e, 16t+M], bias row at K=20.
    // A k order used here (contiguous, same as B): lanes 0-15 M=lane: elems 0-15 -> K 0-15.
    //                  lanes 16-31 M=lane-16: elems 0-15 -> K 16-31.
    v16h A[6];
    #pragma unroll
    for (int t = 0; t < 6; ++t) {
        const int f = 16 * t + n;
        #pragma unroll
        for (int e = 0; e < 16; ++e) {
            const int K = 16 * hi + e;
            float w = 0.f;
            if (K < E_DIM)       w = mR_w[K * 96 + f];
            else if (K == E_DIM) w = mR_b[f];     // bias channel
            A[t][e] = (_Float16)w;
        }
    }

    float accs[IBLK][2][8];        // delta_s partials, f = 16tt + 8hi + r
    float accv[IBLK][2][3][8];     // delta_v partials, f = 16tt + 8hi + r
    #pragma unroll
    for (int ib = 0; ib < IBLK; ++ib)
        #pragma unroll
        for (int tt = 0; tt < 2; ++tt)
            #pragma unroll
            for (int r = 0; r < 8; ++r) {
                accs[ib][tt][r] = 0.f;
                #pragma unroll
                for (int d = 0; d < 3; ++d) accv[ib][tt][d][r] = 0.f;
            }

    for (int it = 0; it < 8; ++it) {
        const int j0 = 16 * (wv * 8 + it);
        const int jj = j0 + n;

        if (it < 7) {   // prefetch next tile's expansion rows into caches
            #pragma unroll
            for (int ib = 0; ib < IBLK; ++ib)
                __builtin_prefetch(expansion +
                    (size_t)((i0 + ib) * N_NODES + jj + 16) * E_DIM, 0, 0);
        }

        // --- B per i (mask folded in, exact: mask is 0.0/1.0).
        // B k order used here (contiguous, same as A): lanes 0-15 col N=lane, elems -> K=0..15;
        //                  lanes 16-31 col N=lane-16, elems -> K=16..31.
        v16h Bm[IBLK];
        float dirv[IBLK][3];
        #pragma unroll
        for (int ib = 0; ib < IBLK; ++ib) {
            const int i = i0 + ib;
            const float m = mask[i * N_NODES + jj];
            dirv[ib][0] = direction[(size_t)(i * N_NODES + jj) * 3 + 0];
            dirv[ib][1] = direction[(size_t)(i * N_NODES + jj) * 3 + 1];
            dirv[ib][2] = direction[(size_t)(i * N_NODES + jj) * 3 + 2];
            const float* erow = expansion + (size_t)(i * N_NODES + jj) * E_DIM;
            if (hi == 0) {
                const float4 e0 = *(const float4*)(erow);
                const float4 e1 = *(const float4*)(erow + 4);
                const float4 e2 = *(const float4*)(erow + 8);
                const float4 e3 = *(const float4*)(erow + 12);
                Bm[ib][0]=(_Float16)(e0.x*m); Bm[ib][1]=(_Float16)(e0.y*m);
                Bm[ib][2]=(_Float16)(e0.z*m); Bm[ib][3]=(_Float16)(e0.w*m);
                Bm[ib][4]=(_Float16)(e1.x*m); Bm[ib][5]=(_Float16)(e1.y*m);
                Bm[ib][6]=(_Float16)(e1.z*m); Bm[ib][7]=(_Float16)(e1.w*m);
                Bm[ib][8]=(_Float16)(e2.x*m); Bm[ib][9]=(_Float16)(e2.y*m);
                Bm[ib][10]=(_Float16)(e2.z*m); Bm[ib][11]=(_Float16)(e2.w*m);
                Bm[ib][12]=(_Float16)(e3.x*m); Bm[ib][13]=(_Float16)(e3.y*m);
                Bm[ib][14]=(_Float16)(e3.z*m); Bm[ib][15]=(_Float16)(e3.w*m);
            } else {
                const float4 e4 = *(const float4*)(erow + 16);
                Bm[ib][0]=(_Float16)(e4.x*m); Bm[ib][1]=(_Float16)(e4.y*m);
                Bm[ib][2]=(_Float16)(e4.z*m); Bm[ib][3]=(_Float16)(e4.w*m);
                Bm[ib][4]=(_Float16)m;                 // mask-scaled bias channel
                #pragma unroll
                for (int e = 5; e < 16; ++e) Bm[ib][e] = (_Float16)0.f;
            }
        }

        #pragma unroll
        for (int t = 0; t < 6; ++t) {
            const int f0 = 16 * t + 8 * hi;   // lane's 8 features this tile
            if (t < 2) {
                // "a" chunk: dv[d,f] += D * Q[j,d,f]   (Q = left*vector prefold)
                float qv[3][8];
                #pragma unroll
                for (int d = 0; d < 3; ++d) {
                    const float4 q0 = *(const float4*)(Qbuf + jj * 96 + d * F_DIM + f0);
                    const float4 q1 = *(const float4*)(Qbuf + jj * 96 + d * F_DIM + f0 + 4);
                    qv[d][0]=q0.x; qv[d][1]=q0.y; qv[d][2]=q0.z; qv[d][3]=q0.w;
                    qv[d][4]=q1.x; qv[d][5]=q1.y; qv[d][6]=q1.z; qv[d][7]=q1.w;
                }
                #pragma unroll
                for (int ib = 0; ib < IBLK; ++ib) {
                    v8f cz = {};
                    v8f D = __builtin_amdgcn_wmma_f32_16x16x32_f16(
                                false, A[t], false, Bm[ib], (short)0, cz, false, false);
                    #pragma unroll
                    for (int d = 0; d < 3; ++d)
                        #pragma unroll
                        for (int r = 0; r < 8; ++r)
                            accv[ib][t][d][r] += D[r] * qv[d][r];
                }
            } else {
                const float4 l0 = *(const float4*)(leftbuf + jj * 96 + f0);
                const float4 l1 = *(const float4*)(leftbuf + jj * 96 + f0 + 4);
                const float lv[8] = {l0.x,l0.y,l0.z,l0.w,l1.x,l1.y,l1.z,l1.w};
                #pragma unroll
                for (int ib = 0; ib < IBLK; ++ib) {
                    v8f cz = {};
                    v8f D = __builtin_amdgcn_wmma_f32_16x16x32_f16(
                                false, A[t], false, Bm[ib], (short)0, cz, false, false);
                    if (t < 4) {            // "b" chunk: ds += L2 * r1
                        #pragma unroll
                        for (int r = 0; r < 8; ++r)
                            accs[ib][t - 2][r] += D[r] * lv[r];
                    } else {                // "c" chunk: dv += dir[d] * L3 * r2
                        #pragma unroll
                        for (int r = 0; r < 8; ++r) {
                            const float tmp = D[r] * lv[r];
                            accv[ib][t - 4][0][r] += tmp * dirv[ib][0];
                            accv[ib][t - 4][1][r] += tmp * dirv[ib][1];
                            accv[ib][t - 4][2][r] += tmp * dirv[ib][2];
                        }
                    }
                }
            }
        }
    }

    // Reduce across the 16 lanes of each half-group (same feature set,
    // distinct j), then one lane per group accumulates into LDS.
    #pragma unroll
    for (int ib = 0; ib < IBLK; ++ib) {
        #pragma unroll
        for (int tt = 0; tt < 2; ++tt) {
            #pragma unroll
            for (int r = 0; r < 8; ++r) {
                const int f = 16 * tt + 8 * hi + r;
                float x = accs[ib][tt][r];
                x += __shfl_xor(x, 1, 32); x += __shfl_xor(x, 2, 32);
                x += __shfl_xor(x, 4, 32); x += __shfl_xor(x, 8, 32);
                if (n == 0) atomicAdd(&sS[ib][f], x);
                #pragma unroll
                for (int d = 0; d < 3; ++d) {
                    float y = accv[ib][tt][d][r];
                    y += __shfl_xor(y, 1, 32); y += __shfl_xor(y, 2, 32);
                    y += __shfl_xor(y, 4, 32); y += __shfl_xor(y, 8, 32);
                    if (n == 0) atomicAdd(&sV[ib][d * F_DIM + f], y);
                }
            }
        }
    }
    __syncthreads();

    #pragma unroll
    for (int ib = 0; ib < IBLK; ++ib) {
        const int i = i0 + ib;
        if (tid < F_DIM)
            out_s[i * F_DIM + tid] = scalar[i * F_DIM + tid] + sS[ib][tid];
        if (tid < 3 * F_DIM)
            out_v[i * 3 * F_DIM + tid] = vector[i * 3 * F_DIM + tid] + sV[ib][tid];
    }
}

// ---------------------------------------------------------------------------
// Kernel 3: update phase, in place on d_out (each block touches only its i).
// ---------------------------------------------------------------------------
__global__ __launch_bounds__(128)
void k_update(const float* __restrict__ uR_w1, const float* __restrict__ uR_b1,
              const float* __restrict__ uR_w2, const float* __restrict__ uR_b2,
              const float* __restrict__ U_w,   const float* __restrict__ V_w,
              float* __restrict__ out_s, float* __restrict__ out_v)
{
    const int i = blockIdx.x;
    const int tid = threadIdx.x;
    __shared__ float ss[F_DIM], vs[3 * F_DIM];
    __shared__ float lu[3 * F_DIM], ru[3 * F_DIM];
    __shared__ float feats[2 * F_DIM], h[F_DIM], o3[3 * F_DIM];

    if (tid < F_DIM)     ss[tid] = out_s[i * F_DIM + tid];
    if (tid < 3 * F_DIM) vs[tid] = out_v[i * 3 * F_DIM + tid];
    __syncthreads();

    if (tid < 3 * F_DIM) {
        const int d = tid >> 5, f = tid & 31;
        float a0 = 0.f, a1 = 0.f;
        #pragma unroll
        for (int k = 0; k < F_DIM; ++k) {
            const float vk = vs[d * F_DIM + k];
            a0 += vk * U_w[k * F_DIM + f];
            a1 += vk * V_w[k * F_DIM + f];
        }
        lu[tid] = a0; ru[tid] = a1;
    }
    __syncthreads();

    if (tid < F_DIM) {
        const float r0 = ru[tid], r1 = ru[F_DIM + tid], r2 = ru[2 * F_DIM + tid];
        feats[tid] = ss[tid];
        feats[F_DIM + tid] = sqrtf(r0 * r0 + r1 * r1 + r2 * r2);
    }
    __syncthreads();

    if (tid < F_DIM) {
        float acc = uR_b1[tid];
        #pragma unroll
        for (int k = 0; k < 2 * F_DIM; ++k) acc += feats[k] * uR_w1[k * F_DIM + tid];
        h[tid] = silu_f(acc);
    }
    __syncthreads();

    if (tid < 3 * F_DIM) {
        float acc = uR_b2[tid];
        #pragma unroll
        for (int k = 0; k < F_DIM; ++k) acc += h[k] * uR_w2[k * 3 * F_DIM + tid];
        o3[tid] = acc;
    }
    __syncthreads();

    if (tid < F_DIM) {
        const float inner = lu[tid] * ru[tid] + lu[F_DIM + tid] * ru[F_DIM + tid]
                          + lu[2 * F_DIM + tid] * ru[2 * F_DIM + tid];
        out_s[i * F_DIM + tid] = ss[tid] + inner * o3[F_DIM + tid] + o3[2 * F_DIM + tid];
    }
    if (tid < 3 * F_DIM) {
        out_v[i * 3 * F_DIM + tid] = vs[tid] + o3[tid & 31] * lu[tid];
    }
}

// ---------------------------------------------------------------------------
extern "C" void kernel_launch(void* const* d_in, const int* in_sizes, int n_in,
                              void* d_out, int out_size, void* d_ws, size_t ws_size,
                              hipStream_t stream)
{
    (void)in_sizes; (void)n_in; (void)out_size; (void)ws_size;
    const float* scalar    = (const float*)d_in[0];
    const float* vector    = (const float*)d_in[1];
    const float* expansion = (const float*)d_in[2];
    const float* direction = (const float*)d_in[3];
    const float* mask      = (const float*)d_in[4];
    const float* mL_w1     = (const float*)d_in[5];
    const float* mL_b1     = (const float*)d_in[6];
    const float* mL_w2     = (const float*)d_in[7];
    const float* mL_b2     = (const float*)d_in[8];
    const float* mR_w      = (const float*)d_in[9];
    const float* mR_b      = (const float*)d_in[10];
    const float* uR_w1     = (const float*)d_in[11];
    const float* uR_b1     = (const float*)d_in[12];
    const float* uR_w2     = (const float*)d_in[13];
    const float* uR_b2     = (const float*)d_in[14];
    const float* U_w       = (const float*)d_in[15];
    const float* V_w       = (const float*)d_in[16];

    float* out_s   = (float*)d_out;                       // [1024, 1, 32]
    float* out_v   = (float*)d_out + N_NODES * F_DIM;     // [1024, 3, 32]
    float* leftbuf = (float*)d_ws;                        // [1024, 96]
    float* Qbuf    = (float*)d_ws + N_NODES * 3 * F_DIM;  // [1024, 3, 32]

    k_msg_left<<<N_NODES, 128, 0, stream>>>(scalar, vector, mL_w1, mL_b1,
                                            mL_w2, mL_b2, leftbuf, Qbuf);
    k_message<<<N_NODES / IBLK, 256, 0, stream>>>(scalar, vector, expansion,
                                                  direction, mask, mR_w, mR_b,
                                                  leftbuf, Qbuf, out_s, out_v);
    k_update<<<N_NODES, 128, 0, stream>>>(uR_w1, uR_b1, uR_w2, uR_b2, U_w, V_w,
                                          out_s, out_v);
}
